// TemporalGCN_54949811585620
// MI455X (gfx1250) — matrix operand order repaired and checked by reading, not compiled, not run on hardware
//
#include <hip/hip_runtime.h>
#include <hip/hip_bf16.h>
#include <math.h>

// ---------------------------------------------------------------------------
// TemporalGCN for MI455X (gfx1250, wave32, WMMA + TDM).
// Heavy op: per-t adj^T[2048x2048] @ Z[2048x64] as bf16 WMMA
// (v_wmma_f32_16x16x32_bf16). adj converted once to transposed bf16 (134 MB,
// L2-resident in MI455X's 192MB L2). B tiles staged into double-buffered LDS
// by the Tensor Data Mover (tensor_load_to_lds, TENSORcnt), overlapping DMA
// with the WMMA stream; one workgroup barrier per K-chunk.
// ---------------------------------------------------------------------------

#define TT 16
#define NN 2048
#define HID 64
#define OUT 32

typedef __attribute__((ext_vector_type(16))) __bf16 v16bf;
typedef __attribute__((ext_vector_type(8)))  __bf16 v8bf;
typedef __attribute__((ext_vector_type(8)))  float  v8f;
typedef unsigned int u32x4 __attribute__((ext_vector_type(4)));
typedef int          i32x8 __attribute__((ext_vector_type(8)));
typedef int          i32x4 __attribute__((ext_vector_type(4)));

#if __has_builtin(__builtin_amdgcn_tensor_load_to_lds)
#define HAVE_TDM 1
#else
#define HAVE_TDM 0
#endif

__device__ __forceinline__ unsigned short f2bf(float f) {
    unsigned int u = __float_as_uint(f);
    unsigned int lsb = (u >> 16) & 1u;
    u += 0x7fffu + lsb;                // round-to-nearest-even
    return (unsigned short)(u >> 16);
}

__device__ __forceinline__ float sigm(float x) {
    return 1.0f / (1.0f + __expf(-x));
}

#if HAVE_TDM
// TDM: DMA a 64-row x 32-elem bf16 tile (row stride 2048 elems) into LDS.
// D# per CDNA5 ISA ch.8: group0 {count, lds_addr, global_addr, type=2},
// group1 {data_size=2B, tensor dims (large, never clip), tile 32x64,
// dim0_stride=2048}. Groups 2/3 zero (<=2D tile).
// clang-23 toolchain: 6-arg form (extra int32x8 before cpol).
__device__ __forceinline__ void tdm_load_b_tile(unsigned lds_off,
                                                const unsigned short* gsrc) {
    unsigned long long ga = (unsigned long long)(uintptr_t)gsrc;
    u32x4 g0 = { 1u,                                  // count=1, user desc
                 lds_off,                             // lds_addr (bytes)
                 (unsigned)(ga & 0xffffffffull),      // global_addr[31:0]
                 (unsigned)((ga >> 32) & 0x1ffffffull) | 0x80000000u }; // +type=2
    i32x8 g1 = { 0x00010000,                          // data_size=1 (2 bytes)
                 0,                                   // tensor_dim0[15:0]=0
                 0x10,                                // tensor_dim0=1<<20; td1 lo=0
                 (int)((32u << 16) | 0x10u),          // tile_dim0=32; td1=1<<20
                 64,                                  // tile_dim1=64, tile_dim2=0
                 2048,                                // tensor_dim0_stride lo
                 0, 0 };
    i32x4 gz  = { 0, 0, 0, 0 };
    i32x8 gz8 = { 0, 0, 0, 0, 0, 0, 0, 0 };
    __builtin_amdgcn_tensor_load_to_lds(g0, g1, gz, gz, gz8, 0);
}
#endif

// ---------------------------------------------------------------------------
// mf[t][n] from ego[B][T][MAXN]; also zero colsum (no memset needed).
// ---------------------------------------------------------------------------
__global__ void k_mask(const unsigned char* __restrict__ ego,
                       float* __restrict__ mf, float* __restrict__ colsum) {
    int idx = blockIdx.x * 256 + threadIdx.x;       // T*N
    int t = idx >> 11, n = idx & (NN - 1);
    int b = n >> 8, j = n & 255;
    mf[idx] = ego[(b * TT + t) * 256 + j] ? 1.0f : 0.0f;
    colsum[idx] = 0.0f;
}

// ---------------------------------------------------------------------------
// adj fp32 [t][s][d] -> adjT bf16 [t][d][s]  (LDS-tiled transpose)
// fused masked column sum: colsum[t][d] += sum_s adj[s][d]*mf[t][s]
// ---------------------------------------------------------------------------
__global__ void k_convert(const float* __restrict__ adj,
                          const float* __restrict__ mf,
                          unsigned short* __restrict__ adjT,
                          float* __restrict__ colsum) {
    __shared__ float tile[64][65];
    const int t = blockIdx.z;
    const int d0 = blockIdx.x * 64;
    const int s0 = blockIdx.y * 64;
    const float* base = adj + (size_t)t * NN * NN;
    const int tx = threadIdx.x & 63;
    const int ty = threadIdx.x >> 6;                // 4 rows per pass
#pragma unroll
    for (int i = 0; i < 16; ++i) {
        int s = ty + i * 4;
        tile[s][tx] = base[(size_t)(s0 + s) * NN + d0 + tx];
    }
    __syncthreads();
    if (threadIdx.x < 64) {
        float acc = 0.0f;
        const float* mrow = mf + t * NN + s0;
#pragma unroll 8
        for (int s = 0; s < 64; ++s) acc += tile[s][threadIdx.x] * mrow[s];
        atomicAdd(&colsum[t * NN + d0 + threadIdx.x], acc);
    }
#pragma unroll
    for (int i = 0; i < 16; ++i) {
        int dr = ty + i * 4;
        adjT[((size_t)t * NN + d0 + dr) * NN + s0 + tx] = f2bf(tile[tx][dr]);
    }
}

// dinv[t][d]: deg = m*(colsum+1); dinv = deg>0 ? rsqrt : 0
__global__ void k_dinv(const float* __restrict__ mf,
                       const float* __restrict__ colsum,
                       float* __restrict__ dinv) {
    int idx = blockIdx.x * 256 + threadIdx.x;
    float m = mf[idx];
    float deg = m * (colsum[idx] + 1.0f);
    dinv[idx] = (deg > 0.0f) ? rsqrtf(deg) : 0.0f;
}

// ---------------------------------------------------------------------------
// Layer-1 feature build: XW[t][n][h] = x@W1 ; ZT[t][h][n] = bf16(dinv*XW)
// ---------------------------------------------------------------------------
__global__ void k_zbuild1(const float* __restrict__ x,
                          const float* __restrict__ W1,
                          const float* __restrict__ dinv,
                          float* __restrict__ XW,
                          unsigned short* __restrict__ ZT) {
    const int t = blockIdx.y;
    const int idx = blockIdx.x * 256 + threadIdx.x;   // N*HID
    const int n = idx >> 6, h = idx & 63;
    float x0 = x[((size_t)t * NN + n) * 2 + 0];
    float x1 = x[((size_t)t * NN + n) * 2 + 1];
    float xw = x0 * W1[h] + x1 * W1[HID + h];
    XW[((size_t)t * NN + n) * HID + h] = xw;
    ZT[((size_t)t * HID + h) * NN + n] = f2bf(dinv[t * NN + n] * xw);
}

// ---------------------------------------------------------------------------
// Layer-2 feature build: XW2 = h1 @ W2 ; ZT2 = bf16(dinv*XW2)
// ---------------------------------------------------------------------------
__global__ void k_zbuild2(const float* __restrict__ h1,
                          const float* __restrict__ W2,
                          const float* __restrict__ dinv,
                          float* __restrict__ XW,
                          unsigned short* __restrict__ ZT) {
    const int t = blockIdx.y;
    const int n0 = blockIdx.x * 4;                    // 4 nodes / block
    __shared__ float lh[4 * HID];
    lh[threadIdx.x] = h1[((size_t)t * NN + n0) * HID + threadIdx.x];
    __syncthreads();
    const int nl = threadIdx.x >> 6, h = threadIdx.x & 63;
    float acc = 0.0f;
#pragma unroll 8
    for (int k = 0; k < HID; ++k) acc += lh[nl * HID + k] * W2[k * HID + h];
    const int n = n0 + nl;
    XW[((size_t)t * NN + n) * HID + h] = acc;
    ZT[((size_t)t * HID + h) * NN + n] = f2bf(dinv[t * NN + n] * acc);
}

// ---------------------------------------------------------------------------
// WMMA GEMM: out[t][d][h] = dinv[d]*(adjT[d,:]·Z[:,h]) + dinv[d]^2*XW[d][h] + b[h]
// mode 1: relu (layer 1, writes h1); mode 2: *mf (layer 2, writes h2)
// Block: 256 thr (8 waves). Wave w owns M-tile w (16 dst rows), all 4 N-tiles.
// K-chunks of 32. B tile DMA'd into double-buffered LDS by the TDM (wave 0
// issues tensor_load_to_lds; s_wait_tensorcnt + one barrier per chunk).
// A loaded directly from L2-resident transposed bf16 adjacency.
// ---------------------------------------------------------------------------
__global__ void k_gemm(const unsigned short* __restrict__ adjT,
                       const unsigned short* __restrict__ ZT,
                       const float* __restrict__ XW,
                       const float* __restrict__ dinv,
                       const float* __restrict__ bias,
                       const float* __restrict__ mf,
                       float* __restrict__ out, int mode) {
    const int t    = blockIdx.y;
    const int dblk = blockIdx.x * 128;
    const int wave = threadIdx.x >> 5;
    const int lane = threadIdx.x & 31;
    const int half = lane >> 4;
    const int lm   = lane & 15;

    __shared__ __align__(32) unsigned short ldsB[2][HID * 32];   // 2 x 4 KB

    const unsigned short* Abase =
        adjT + ((size_t)t * NN + dblk + wave * 16 + lm) * NN;
    const unsigned short* Bbase = ZT + (size_t)t * HID * NN;

    const v8f vz = {0.f, 0.f, 0.f, 0.f, 0.f, 0.f, 0.f, 0.f};
    v8f acc[4] = {vz, vz, vz, vz};

#if HAVE_TDM
    const unsigned lds_off0 = (unsigned)(uintptr_t)&ldsB[0][0];
    const unsigned lds_off1 = (unsigned)(uintptr_t)&ldsB[1][0];
    if (wave == 0) {
        tdm_load_b_tile(lds_off0, Bbase);          // prologue: chunk 0 -> buf0
        __builtin_amdgcn_s_wait_tensorcnt(0);
    }
    __syncthreads();
    for (int i = 0; i < NN / 32; ++i) {
        const int k0 = i * 32;
        if (wave == 0 && i + 1 < NN / 32)          // DMA next chunk -> other buf
            tdm_load_b_tile((i & 1) ? lds_off0 : lds_off1, Bbase + k0 + 32);

        if (k0 + 32 < NN)                          // near-scope prefetch of next A
            __builtin_prefetch(Abase + k0 + 32 + 8 * half, 0, 3);

        // A fragment (matches B): lane(m=lm,half) holds K = 16*half + 0..15 contiguous
        v16bf a = *(const v16bf*)(Abase + k0 + 16 * half);
#pragma unroll
        for (int nt = 0; nt < 4; ++nt) {
            // B fragment: lane(n=lm,half) holds K = 16*half + 0..15 contiguous
            v16bf b = *(const v16bf*)(&ldsB[i & 1][(nt * 16 + lm) * 32 + 16 * half]);
            acc[nt] = __builtin_amdgcn_wmma_f32_16x16x32_bf16(
                false, a, false, b, (short)0, acc[nt], false, false);
        }
        if (wave == 0) __builtin_amdgcn_s_wait_tensorcnt(0);
        __syncthreads();
    }
#else
    // Fallback: cooperative vector staging (single-buffer), 2 barriers/chunk.
    for (int k0 = 0; k0 < NN; k0 += 32) {
        {
            const int row  = threadIdx.x >> 2;
            const int part = threadIdx.x & 3;
            v8bf src = *(const v8bf*)(Bbase + (size_t)row * NN + k0 + part * 8);
            *(v8bf*)(&ldsB[0][row * 32 + part * 8]) = src;
        }
        __syncthreads();
        if (k0 + 32 < NN)
            __builtin_prefetch(Abase + k0 + 32 + 8 * half, 0, 3);
        v16bf a = *(const v16bf*)(Abase + k0 + 16 * half);
#pragma unroll
        for (int nt = 0; nt < 4; ++nt) {
            v16bf b = *(const v16bf*)(&ldsB[0][(nt * 16 + lm) * 32 + 16 * half]);
            acc[nt] = __builtin_amdgcn_wmma_f32_16x16x32_bf16(
                false, a, false, b, (short)0, acc[nt], false, false);
        }
        __syncthreads();
    }
#endif

    // epilogue: C layout -> elem r: M = r + 8*half, N = lm
    const int dw = dblk + wave * 16;
#pragma unroll
    for (int nt = 0; nt < 4; ++nt) {
        const int h = nt * 16 + lm;
        const float bh = bias[h];
#pragma unroll
        for (int r = 0; r < 8; ++r) {
            const int d = dw + r + 8 * half;
            const float di = dinv[t * NN + d];
            float val = di * acc[nt][r]
                      + di * di * XW[((size_t)t * NN + d) * HID + h] + bh;
            if (mode == 1) val = fmaxf(val, 0.0f);
            else           val *= mf[t * NN + d];
            out[((size_t)t * NN + d) * HID + h] = val;
        }
    }
}

// ---------------------------------------------------------------------------
// LSTM over node axis (2048 sequential steps), batch = 16 timesteps.
// One block per batch element; 128 threads = one gate row each.
// ---------------------------------------------------------------------------
__global__ void k_lstm(const float* __restrict__ h2,
                       const float* __restrict__ W_ih,
                       const float* __restrict__ W_hh,
                       const float* __restrict__ b_ih,
                       const float* __restrict__ b_hh,
                       float* __restrict__ lstm_out) {
    const int tb = blockIdx.x;
    const int j = threadIdx.x;                       // gate row 0..127
    float wih[HID], whh[OUT];
#pragma unroll 8
    for (int k = 0; k < HID; ++k) wih[k] = W_ih[j * HID + k];
#pragma unroll 8
    for (int k = 0; k < OUT; ++k) whh[k] = W_hh[j * OUT + k];
    const float bsum = b_ih[j] + b_hh[j];

    __shared__ float lx[HID], lh[OUT], lc[OUT], lg[128];
    if (j < OUT) { lh[j] = 0.0f; lc[j] = 0.0f; }
    __syncthreads();

    for (int n = 0; n < NN; ++n) {
        if (j < HID) lx[j] = h2[((size_t)tb * NN + n) * HID + j];
        __syncthreads();
        float g = bsum;
#pragma unroll 8
        for (int k = 0; k < HID; ++k) g += wih[k] * lx[k];
#pragma unroll 8
        for (int k = 0; k < OUT; ++k) g += whh[k] * lh[k];
        lg[j] = g;
        __syncthreads();
        if (j < OUT) {
            float gi = lg[j], gf = lg[j + 32], gg = lg[j + 64], go = lg[j + 96];
            float c = sigm(gf) * lc[j] + sigm(gi) * tanhf(gg);
            float hv = sigm(go) * tanhf(c);
            lc[j] = c; lh[j] = hv;
            lstm_out[((size_t)n * TT + tb) * OUT + j] = hv;
        }
        __syncthreads();
    }
}

// ---------------------------------------------------------------------------
// softmax attention over T per node + weighted pool -> d_out[n*32+h]
// ---------------------------------------------------------------------------
__global__ void k_attnpool(const float* __restrict__ lstm_out,
                           const float* __restrict__ Wa,
                           const float* __restrict__ ba,
                           float* __restrict__ out) {
    const int n = blockIdx.x * blockDim.x + threadIdx.x;
    if (n >= NN) return;
    const float* base = lstm_out + (size_t)n * TT * OUT;
    float sc[TT], mx = -1e30f;
#pragma unroll
    for (int t = 0; t < TT; ++t) {
        float s = ba[0];
#pragma unroll 8
        for (int h = 0; h < OUT; ++h) s += base[t * OUT + h] * Wa[h];
        sc[t] = s;
        mx = fmaxf(mx, s);
    }
    float den = 0.0f;
#pragma unroll
    for (int t = 0; t < TT; ++t) { sc[t] = __expf(sc[t] - mx); den += sc[t]; }
    const float inv = 1.0f / den;
#pragma unroll 8
    for (int h = 0; h < OUT; ++h) {
        float p = 0.0f;
#pragma unroll
        for (int t = 0; t < TT; ++t) p += base[t * OUT + h] * sc[t];
        out[(size_t)n * OUT + h] = p * inv;
    }
}

// ---------------------------------------------------------------------------
extern "C" void kernel_launch(void* const* d_in, const int* in_sizes, int n_in,
                              void* d_out, int out_size, void* d_ws, size_t ws_size,
                              hipStream_t stream) {
    const float* x    = (const float*)d_in[0];            // [T,N,2]
    const float* adj  = (const float*)d_in[1];            // [T,N,N]
    const unsigned char* ego = (const unsigned char*)d_in[2]; // [B,T,256] bool
    const float* W1   = (const float*)d_in[3];
    const float* b1   = (const float*)d_in[4];
    const float* W2   = (const float*)d_in[5];
    const float* b2   = (const float*)d_in[6];
    const float* W_ih = (const float*)d_in[7];
    const float* W_hh = (const float*)d_in[8];
    const float* b_ih = (const float*)d_in[9];
    const float* b_hh = (const float*)d_in[10];
    const float* Wa   = (const float*)d_in[11];
    const float* ba   = (const float*)d_in[12];
    float* outp = (float*)d_out;

    // workspace carve (~167 MB total)
    char* p = (char*)d_ws;
    auto carve = [&](size_t bytes) -> void* {
        void* r = (void*)p;
        p += (bytes + 255) & ~(size_t)255;
        return r;
    };
    unsigned short* adjT = (unsigned short*)carve((size_t)TT * NN * NN * 2); // 134 MB, L2-resident
    float* mf      = (float*)carve((size_t)TT * NN * 4);
    float* colsum  = (float*)carve((size_t)TT * NN * 4);
    float* dinv    = (float*)carve((size_t)TT * NN * 4);
    unsigned short* ZT = (unsigned short*)carve((size_t)TT * HID * NN * 2);
    float* XW      = (float*)carve((size_t)TT * NN * HID * 4);
    float* h1      = (float*)carve((size_t)TT * NN * HID * 4);
    float* h2      = (float*)carve((size_t)TT * NN * HID * 4);
    float* lstm_o  = (float*)carve((size_t)NN * TT * OUT * 4);
    (void)ws_size; (void)n_in; (void)in_sizes; (void)out_size;

    // 1. mask + colsum zero
    k_mask<<<(TT * NN) / 256, 256, 0, stream>>>(ego, mf, colsum);
    // 2. adj -> transposed bf16 + masked degrees
    k_convert<<<dim3(NN / 64, NN / 64, TT), 256, 0, stream>>>(adj, mf, adjT, colsum);
    // 3. dinv
    k_dinv<<<(TT * NN) / 256, 256, 0, stream>>>(mf, colsum, dinv);
    // 4. layer-1 features
    k_zbuild1<<<dim3((NN * HID) / 256, TT), 256, 0, stream>>>(x, W1, dinv, XW, ZT);
    // 5. GCN layer 1 (WMMA + TDM) -> h1 (relu)
    k_gemm<<<dim3(NN / 128, TT), 256, 0, stream>>>(adjT, ZT, XW, dinv, b1, mf, h1, 1);
    // 6. layer-2 features
    k_zbuild2<<<dim3(NN / 4, TT), 256, 0, stream>>>(h1, W2, dinv, XW, ZT);
    // 7. GCN layer 2 (WMMA + TDM) -> h2 (masked)
    k_gemm<<<dim3(NN / 128, TT), 256, 0, stream>>>(adjT, ZT, XW, dinv, b2, mf, h2, 2);
    // 8. LSTM scan over nodes
    k_lstm<<<TT, 128, 0, stream>>>(h2, W_ih, W_hh, b_ih, b_hh, lstm_o);
    // 9. attention pool -> output [B,256,OUT]
    k_attnpool<<<NN / 256, 256, 0, stream>>>(lstm_o, Wa, ba, outp);
}
